// VectorQuantizer_30657476559293
// MI455X (gfx1250) — matrix operand order repaired and checked by reading, not compiled, not run on hardware
//
#include <hip/hip_runtime.h>

// ---------------------------------------------------------------------------
// VQ: codes = argmin_k ||x - e_k||^2 ; code_vecs = e[codes]
// B=32, M=32 -> N_ROWS=32768 vectors, DIM=256, K_CODES=8192, all fp32.
// Strategy: bf16 WMMA (v_wmma_f32_16x16x32_bf16) for the 32768x8192x256 dot,
// fp32 norms added around it, per-lane running argmin + cross-lane reduce.
// LDS staging uses GLOBAL_LOAD_ASYNC_TO_LDS_B128 (ASYNCcnt) when available.
// ---------------------------------------------------------------------------

#define N_ROWS  32768
#define K_CODES 8192
#define DIM     256
#define CHUNK   64                    // codes per LDS chunk
#define NCHUNK  (K_CODES / CHUNK)     // 128
#define CHUNK_U4 2048                 // 64 codes * 256 dims * 2B / 16B

#if defined(__AMDGCN__) && __has_builtin(__builtin_amdgcn_global_load_async_to_lds_b128)
#define VQ_USE_ASYNC 1
#else
#define VQ_USE_ASYNC 0
#endif

typedef __attribute__((ext_vector_type(16))) __bf16 v16bf;
typedef __attribute__((ext_vector_type(8)))  float  v8f;

union AFrag { unsigned short h[16]; v16bf v; };
union BFrag { uint4 q[2];           v16bf v; };
union Pack8 { unsigned short h[8];  uint4 q; };

__device__ __forceinline__ unsigned short f2bf(float f) {
  // round-to-nearest-even fp32 -> bf16
  unsigned int u = __float_as_uint(f);
  u += 0x7FFFu + ((u >> 16) & 1u);
  return (unsigned short)(u >> 16);
}

#if VQ_USE_ASYNC
// Builtin parameter types: (v4i addrspace(1)*, v4i addrspace(3)*, imm, imm)
typedef int v4i __attribute__((vector_size(16)));
typedef __attribute__((address_space(1))) v4i* gv4i_p;
typedef __attribute__((address_space(3))) v4i* lv4i_p;

__device__ __forceinline__ void vq_wait_async0() {
#if __has_builtin(__builtin_amdgcn_s_wait_asynccnt)
  __builtin_amdgcn_s_wait_asynccnt(0);
#else
  asm volatile("s_wait_asynccnt 0" ::: "memory");
#endif
}
#endif

// ---------------------------------------------------------------------------
// Kernel 1: embeddings fp32 -> bf16 tiled table (B-frag friendly) + ||e||^2.
// Tiled layout (uint4 units): idx = (code/64)*2048 + d8*64 + (code%64),
// where d8 = dim/8 (16B of bf16 = 8 dims). One wave per code, lane = d8.
// ---------------------------------------------------------------------------
__global__ __launch_bounds__(256) void vq_prep_embed(
    const float* __restrict__ emb, uint4* __restrict__ wsb4,
    float* __restrict__ enorm) {
  int wave = threadIdx.x >> 5, lane = threadIdx.x & 31;
  int code = blockIdx.x * 8 + wave;                 // grid = 1024
  const float* p = emb + code * DIM + lane * 8;
  float4 f0 = *(const float4*)(p);
  float4 f1 = *(const float4*)(p + 4);
  float s = f0.x*f0.x + f0.y*f0.y + f0.z*f0.z + f0.w*f0.w
          + f1.x*f1.x + f1.y*f1.y + f1.z*f1.z + f1.w*f1.w;
  Pack8 pk;
  pk.h[0]=f2bf(f0.x); pk.h[1]=f2bf(f0.y); pk.h[2]=f2bf(f0.z); pk.h[3]=f2bf(f0.w);
  pk.h[4]=f2bf(f1.x); pk.h[5]=f2bf(f1.y); pk.h[6]=f2bf(f1.z); pk.h[7]=f2bf(f1.w);
  wsb4[(code >> 6) * CHUNK_U4 + lane * 64 + (code & 63)] = pk.q;
#pragma unroll
  for (int m = 16; m >= 1; m >>= 1) s += __shfl_xor(s, m, 32);
  if (lane == 0) enorm[code] = s;
}

// ---------------------------------------------------------------------------
// Kernel 2: ||x||^2 per row. One wave per row, lane covers 8 dims.
// ---------------------------------------------------------------------------
__global__ __launch_bounds__(256) void vq_prep_xnorm(
    const float* __restrict__ x, float* __restrict__ xnorm) {
  int wave = threadIdx.x >> 5, lane = threadIdx.x & 31;
  int row = blockIdx.x * 8 + wave;                  // grid = 4096
  const float* p = x + (size_t)row * DIM + lane * 8;
  float4 f0 = *(const float4*)(p);
  float4 f1 = *(const float4*)(p + 4);
  float s = f0.x*f0.x + f0.y*f0.y + f0.z*f0.z + f0.w*f0.w
          + f1.x*f1.x + f1.y*f1.y + f1.z*f1.z + f1.w*f1.w;
#pragma unroll
  for (int m = 16; m >= 1; m >>= 1) s += __shfl_xor(s, m, 32);
  if (lane == 0) xnorm[row] = s;
}

// ---------------------------------------------------------------------------
// Kernel 3: main search. Block = 256 threads = 8 waves = 256 rows.
// Wave holds its 32 rows' bf16 A operand fully in registers (2 Mtiles x 8
// Ksteps x 8 VGPRs = 128 VGPRs). Codes stream through double-buffered LDS.
// ---------------------------------------------------------------------------
__global__ __launch_bounds__(256) void vq_main(
    const float* __restrict__ x, const uint4* __restrict__ wsb4,
    const float* __restrict__ enorm, const float* __restrict__ xnorm,
    int* __restrict__ codes) {
  __shared__ uint4 lds[2][CHUNK_U4];                // 64 KB double buffer

  const int t = threadIdx.x;
  const int w = t >> 5, l = t & 31;
  const int lrow = l & 15, lhalf = l >> 4;
  const int rbase = blockIdx.x * 256 + w * 32;      // grid = 128

  // ---- Load + convert A: 16-bit A-matrix 16x32 layout per WMMA spec.
  // lane 0-15 : M = lane,  regs 0-3 = K k0..k0+7, regs 4-7 = K k0+16..k0+23 (k0 = ks*32)
  // lane 16-31: M = l-16,  same with k0 += 8
  v16bf a[2][8];
#pragma unroll
  for (int rt = 0; rt < 2; ++rt) {
#pragma unroll
    for (int ks = 0; ks < 8; ++ks) {
      const float* p = x + (size_t)(rbase + rt * 16 + lrow) * DIM
                         + ks * 32 + lhalf * 8;
      float4 f0 = *(const float4*)(p);
      float4 f1 = *(const float4*)(p + 4);
      float4 f2 = *(const float4*)(p + 16);
      float4 f3 = *(const float4*)(p + 20);
      AFrag af;
      af.h[0]=f2bf(f0.x);  af.h[1]=f2bf(f0.y);  af.h[2]=f2bf(f0.z);  af.h[3]=f2bf(f0.w);
      af.h[4]=f2bf(f1.x);  af.h[5]=f2bf(f1.y);  af.h[6]=f2bf(f1.z);  af.h[7]=f2bf(f1.w);
      af.h[8]=f2bf(f2.x);  af.h[9]=f2bf(f2.y);  af.h[10]=f2bf(f2.z); af.h[11]=f2bf(f2.w);
      af.h[12]=f2bf(f3.x); af.h[13]=f2bf(f3.y); af.h[14]=f2bf(f3.z); af.h[15]=f2bf(f3.w);
      a[rt][ks] = af.v;
    }
  }

  // ---- x-norms aligned with C-layout rows: lane l, acc reg r -> M = r + 8*(l>>4)
  float xn[2][8];
#pragma unroll
  for (int rt = 0; rt < 2; ++rt)
#pragma unroll
    for (int r = 0; r < 8; ++r)
      xn[rt][r] = xnorm[rbase + rt * 16 + r + 8 * lhalf];

  float mind[2][8];
  int   midx[2][8];
#pragma unroll
  for (int rt = 0; rt < 2; ++rt)
#pragma unroll
    for (int r = 0; r < 8; ++r) { mind[rt][r] = 3.4e38f; midx[rt][r] = 0; }

  // ---- stage chunk 0
#if VQ_USE_ASYNC
  {
#pragma unroll
    for (int i = 0; i < 8; ++i)
      __builtin_amdgcn_global_load_async_to_lds_b128(
          (gv4i_p)(wsb4 + i * 256 + t),
          (lv4i_p)&lds[0][i * 256 + t], 0, 0);
    vq_wait_async0();
  }
#else
#pragma unroll
  for (int i = 0; i < 8; ++i) lds[0][i * 256 + t] = wsb4[i * 256 + t];
#endif
  __syncthreads();

  for (int ch = 0; ch < NCHUNK; ++ch) {
    const int cur = ch & 1;
    const bool has_next = (ch + 1) < NCHUNK;

#if VQ_USE_ASYNC
    // DMA next chunk straight into the other buffer (its last readers
    // finished at the barrier ending iteration ch-1). No VGPR round-trip.
    if (has_next) {
      const uint4* src = wsb4 + (ch + 1) * CHUNK_U4;
#pragma unroll
      for (int i = 0; i < 8; ++i)
        __builtin_amdgcn_global_load_async_to_lds_b128(
            (gv4i_p)(src + i * 256 + t),
            (lv4i_p)&lds[cur ^ 1][i * 256 + t], 0, 0);
    }
#else
    uint4 pf[8];
    if (has_next) {
      const uint4* src = wsb4 + (ch + 1) * CHUNK_U4;
#pragma unroll
      for (int i = 0; i < 8; ++i) pf[i] = src[i * 256 + t];
    }
#endif

    const int cbase = ch * CHUNK;
#pragma unroll
    for (int ct = 0; ct < 4; ++ct) {
      const int code_l = ct * 16 + lrow;            // B column this lane owns
      const float en = enorm[cbase + code_l];
      v8f dot0 = {0.f,0.f,0.f,0.f,0.f,0.f,0.f,0.f};
      v8f dot1 = {0.f,0.f,0.f,0.f,0.f,0.f,0.f,0.f};
#pragma unroll
      for (int ks = 0; ks < 8; ++ks) {
        // B 32x16 layout: lanes 0-15 K=0..7,16..23, lanes 16-31 K=8..15,24..31 of column l%16
        const int d8a = ks * 4 + lhalf;
        BFrag bf;
        bf.q[0] = lds[cur][d8a * 64 + code_l];
        bf.q[1] = lds[cur][(d8a + 2) * 64 + code_l];
        dot0 = __builtin_amdgcn_wmma_f32_16x16x32_bf16(
            false, a[0][ks], false, bf.v, (short)0, dot0, false, false);
        dot1 = __builtin_amdgcn_wmma_f32_16x16x32_bf16(
            false, a[1][ks], false, bf.v, (short)0, dot1, false, false);
      }
      const int idx = cbase + code_l;
#pragma unroll
      for (int r = 0; r < 8; ++r) {
        float d0 = xn[0][r] + en - 2.0f * dot0[r];
        float d1 = xn[1][r] + en - 2.0f * dot1[r];
        if (d0 < mind[0][r]) { mind[0][r] = d0; midx[0][r] = idx; }
        if (d1 < mind[1][r]) { mind[1][r] = d1; midx[1][r] = idx; }
      }
    }

#if VQ_USE_ASYNC
    if (has_next) vq_wait_async0();   // my async stores into lds[cur^1] landed
#else
    if (has_next) {
#pragma unroll
      for (int i = 0; i < 8; ++i) lds[cur ^ 1][i * 256 + t] = pf[i];
    }
#endif
    __syncthreads();
  }

  // ---- cross-lane argmin within each 16-lane half (tie -> lowest index)
#pragma unroll
  for (int rt = 0; rt < 2; ++rt) {
#pragma unroll
    for (int r = 0; r < 8; ++r) {
      float d = mind[rt][r];
      int  ix = midx[rt][r];
#pragma unroll
      for (int m = 1; m <= 8; m <<= 1) {
        float od = __shfl_xor(d, m, 32);
        int   oi = __shfl_xor(ix, m, 32);
        if (od < d || (od == d && oi < ix)) { d = od; ix = oi; }
      }
      if (lrow == 0)
        codes[rbase + rt * 16 + r + 8 * lhalf] = ix;
    }
  }
}

// ---------------------------------------------------------------------------
// Kernel 4: gather code_vecs = embeddings[codes] (fp32, 1 KB/row).
// One wave per row; lane copies 2 float4.
// ---------------------------------------------------------------------------
__global__ __launch_bounds__(256) void vq_gather(
    const int* __restrict__ codes, const float* __restrict__ emb,
    float* __restrict__ out) {
  int wave = threadIdx.x >> 5, lane = threadIdx.x & 31;
  int row = blockIdx.x * 8 + wave;                  // grid = 4096
  int c = codes[row];
  const float4* src = (const float4*)(emb + (size_t)c * DIM);
  float4* dst = (float4*)(out + (size_t)row * DIM);
  dst[lane]      = src[lane];
  dst[lane + 32] = src[lane + 32];
}

// ---------------------------------------------------------------------------
extern "C" void kernel_launch(void* const* d_in, const int* in_sizes, int n_in,
                              void* d_out, int out_size, void* d_ws, size_t ws_size,
                              hipStream_t stream) {
  const float* x   = (const float*)d_in[0];   // [32,32,32,256] fp32
  const float* emb = (const float*)d_in[1];   // [8192,256] fp32

  int*   codes = (int*)d_out;                 // first 32768 outputs = int32 codes
  float* cvecs = (float*)d_out + N_ROWS;      // then 32768*256 fp32 code_vecs

  uint4* wsb4  = (uint4*)d_ws;                                    // 4 MB bf16 table
  float* enorm = (float*)((char*)d_ws + (size_t)K_CODES * DIM * 2);
  float* xnorm = enorm + K_CODES;

  vq_prep_embed<<<K_CODES / 8, 256, 0, stream>>>(emb, wsb4, enorm);
  vq_prep_xnorm<<<N_ROWS / 8, 256, 0, stream>>>(x, xnorm);
  vq_main<<<N_ROWS / 256, 256, 0, stream>>>(x, wsb4, enorm, xnorm, codes);
  vq_gather<<<N_ROWS / 8, 256, 0, stream>>>(codes, emb, cvecs);
}
